// MosLSTM_83648783057402
// MI455X (gfx1250) — matrix operand order repaired and checked by reading, not compiled, not run on hardware
//
#include <hip/hip_runtime.h>

// ---------------- problem constants (match reference) ----------------
#define NEV   16384     // N_EVENTS
#define DIN   128       // D
#define HID   128       // H
#define KPOS  9         // 3x3 kernel positions
#define NSEQ  2048      // S
#define TTOT  147456    // total patch events
#define G4H   512       // 4*H (LSTM gates)

typedef __attribute__((ext_vector_type(16))) _Float16 v16h;
typedef __attribute__((ext_vector_type(8)))  float    v8f;
typedef __attribute__((ext_vector_type(4)))  unsigned u32x4;
typedef __attribute__((ext_vector_type(8)))  unsigned u32x8;

// ---------------- WMMA fragment helpers (A/B use contiguous k = 16*(lane/16) + e) ----------------
__device__ __forceinline__ v16h load_a_frag(const _Float16* src, int ld, int kk, int lane) {
  int m = lane & 15;
  int kbase = (lane < 16) ? 0 : 16;
  const _Float16* p = src + m * ld + kk;
  v16h a;
#pragma unroll
  for (int e = 0; e < 16; ++e) {
    int k = kbase + e;
    a[e] = p[k];
  }
  return a;
}

__device__ __forceinline__ v16h load_b_frag(const _Float16* w, int ld, int n0, int kk, int lane) {
  int n = n0 + (lane & 15);
  int khalf = (lane < 16) ? 0 : 16;
  const _Float16* p = w + n * ld + kk + khalf;
  v16h b;
#pragma unroll
  for (int e = 0; e < 16; ++e) b[e] = p[e];
  return b;
}

__device__ __forceinline__ int cd_row(int lane, int e) { return (lane < 16) ? e : e + 8; }

__device__ __forceinline__ v8f wmma_f16(v16h a, v16h b, v8f c) {
  return __builtin_amdgcn_wmma_f32_16x16x32_f16(false, a, false, b, (short)0, c, false, false);
}

__device__ __forceinline__ float sigmoidf_(float x) { return 1.0f / (1.0f + __expf(-x)); }

// LDS byte offset of a __shared__ object (generic ptr low 32 bits, ISA 10.2)
__device__ __forceinline__ unsigned lds_off_u32(const void* p) {
  return (unsigned)(size_t)p;
}

// ---------------- Tensor Data Mover: 2D fp16 tile, global -> LDS ----------------
// D# per ISA ch.8: group0 = {count, lds_addr, global_addr[56:0], type=2};
// group1 = {data_size=2B, tensor_dim0/1, tile_dim0/1, dim0 stride}.
__device__ __forceinline__ void tdm_load_2d_f16(unsigned lds_byte_off, const void* gptr,
                                                unsigned tensor_d0, unsigned tensor_d1,
                                                unsigned tile_d0, unsigned tile_d1,
                                                unsigned stride0) {
  unsigned long long ga = (unsigned long long)(size_t)gptr;
  u32x4 g0;
  g0[0] = 1u;                                   // count=1, user descriptor
  g0[1] = lds_byte_off;                         // lds_addr
  g0[2] = (unsigned)ga;                         // global_addr[31:0]
  g0[3] = ((unsigned)(ga >> 32) & 0x01FFFFFFu)  // global_addr[56:32]
          | 0x80000000u;                        // type=2 ("image")
  u32x8 g1;
  g1[0] = 0x00010000u;                          // data_size=1 -> 2 bytes
  g1[1] = (tensor_d0 & 0xFFFFu) << 16;          // tensor_dim0[15:0]
  g1[2] = (tensor_d0 >> 16) | ((tensor_d1 & 0xFFFFu) << 16);
  g1[3] = (tensor_d1 >> 16) | (tile_d0 << 16);  // tile_dim0
  g1[4] = tile_d1;                              // tile_dim1 (tile_dim2=0)
  g1[5] = stride0;                              // tensor_dim0_stride[31:0]
  g1[6] = 0u;
  g1[7] = 0u;
  asm volatile("tensor_load_to_lds %0, %1" :: "s"(g0), "s"(g1) : "memory");
}

// Async per-lane 16B copy global -> LDS (GVS form), tracked by ASYNCcnt.
__device__ __forceinline__ void async_load_b128(unsigned lds_byte_off, const void* saddr,
                                                unsigned voffset_bytes) {
  asm volatile("global_load_async_to_lds_b128 %0, %1, %2"
               :: "v"(lds_byte_off), "v"(voffset_bytes), "s"(saddr) : "memory");
}
__device__ __forceinline__ void wait_asynccnt0() {
  asm volatile("s_wait_asynccnt 0x0" ::: "memory");
}

// ---------------- small prep kernels ----------------
__global__ void k_convert(const float* sw, const float* gw, const float* wih, const float* whh,
                          _Float16* sw16, _Float16* gw16, _Float16* wih16, _Float16* whh16) {
  int i = blockIdx.x * blockDim.x + threadIdx.x;
  if (i < KPOS * HID * DIN) { sw16[i] = (_Float16)sw[i]; gw16[i] = (_Float16)gw[i]; }
  if (i < G4H * HID)        { wih16[i] = (_Float16)wih[i]; whh16[i] = (_Float16)whh[i]; }
}

__global__ void k_zero_counts(int* counts) {
  if (threadIdx.x < KPOS) counts[threadIdx.x] = 0;
}

__global__ void k_count(const int* pos_id, int* counts) {
  int t = blockIdx.x * blockDim.x + threadIdx.x;
  if (t < TTOT) atomicAdd(&counts[pos_id[t]], 1);
}

__global__ void k_scan(const int* counts, int* offsets, int* cursors, int* tile_base,
                       const int* lengths, int* seq_off) {
  if (threadIdx.x == 0 && blockIdx.x == 0) {
    int acc = 0, tb = 0;
    for (int k = 0; k < KPOS; ++k) {
      offsets[k] = acc; cursors[k] = acc; tile_base[k] = tb;
      tb  += (counts[k] + 15) >> 4;
      acc += counts[k];
    }
    tile_base[KPOS] = tb;
    int so = 0;
    for (int s = 0; s < NSEQ; ++s) { seq_off[s] = so; so += lengths[s]; }
  }
}

__global__ void k_bin_scatter(const int* pos_id, int* cursors, int* sorted_idx) {
  int t = blockIdx.x * blockDim.x + threadIdx.x;
  if (t < TTOT) {
    int slot = atomicAdd(&cursors[pos_id[t]], 1);
    sorted_idx[slot] = t;
  }
}

__global__ void k_zero_accum(float* accum) {
  int i = blockIdx.x * blockDim.x + threadIdx.x;
  if (i < NEV * HID) accum[i] = 0.0f;
}

// ---------------- stage 1: binned scatter projection (WMMA + TDM) ----------------
__global__ __launch_bounds__(128)
void k_scatter_proj(const float* events, const _Float16* sw16, const int* oe_idx,
                    const int* offsets, const int* counts, const int* tile_base,
                    const int* sorted_idx, _Float16* x16) {
  __shared__ __align__(16) _Float16 xs[16 * DIN];   // in: event tile / out: result tile
  __shared__ __align__(16) _Float16 wsh[HID * DIN]; // W_k row-major [H][D]
  __shared__ int rows[16];
  int bid = blockIdx.x;
  int k = -1, tile = 0;
#pragma unroll
  for (int j = 0; j < KPOS; ++j)
    if (bid >= tile_base[j] && bid < tile_base[j + 1]) { k = j; tile = bid - tile_base[j]; }
  if (k < 0) return;
  int tid = threadIdx.x, wave = tid >> 5, lane = tid & 31;
  int cnt = counts[k], base = offsets[k];
  // TDM: whole 128x128 fp16 W_k as a 1D tile (16384 elems)
  if (wave == 0) {
    tdm_load_2d_f16(lds_off_u32(wsh), sw16 + k * HID * DIN, 16384u, 1u, 16384u, 1u, 16384u);
  }
  if (tid < 16) {
    int idx = tile * 16 + tid;
    rows[tid] = (idx < cnt) ? sorted_idx[base + idx] : -1;
  }
  __syncthreads();
  for (int i = tid; i < 16 * DIN; i += 128) {
    int r = i >> 7, c = i & 127;
    int t = rows[r];
    xs[i] = (_Float16)((t >= 0) ? events[oe_idx[t] * DIN + c] : 0.0f);
  }
  if (wave == 0) __builtin_amdgcn_s_wait_tensorcnt(0);
  __syncthreads();

  v8f accs[2];
#pragma unroll
  for (int nt = 0; nt < 2; ++nt) {                  // 4 waves x 2 tiles = 128 cols
    int n0 = (wave * 2 + nt) * 16;
    v8f acc = {};
#pragma unroll
    for (int kk = 0; kk < DIN; kk += 32) {
      v16h a = load_a_frag(xs, DIN, kk, lane);
      v16h b = load_b_frag(wsh, DIN, n0, kk, lane);
      acc = wmma_f16(a, b, acc);
    }
    accs[nt] = acc;
  }
  __syncthreads();                                  // xs reads done; reuse as out tile
#pragma unroll
  for (int nt = 0; nt < 2; ++nt) {
    int n = (wave * 2 + nt) * 16 + (lane & 15);
#pragma unroll
    for (int e = 0; e < 8; ++e) xs[cd_row(lane, e) * HID + n] = (_Float16)accs[nt][e];
  }
  __syncthreads();
  // row-masked coalesced b128 stores (16 rows x 16 chunks of 16B)
  for (int c = tid; c < 256; c += 128) {
    int m = c >> 4, cb = (c & 15) * 16;
    int t = rows[m];
    if (t >= 0) {
      uint4 v = *(const uint4*)((const char*)&xs[m * HID] + cb);
      *(uint4*)((char*)(x16 + t * HID) + cb) = v;
    }
  }
}

// ---------------- stage 2: hoisted input GEMM xg = x @ Wih^T (WMMA + TDM) ----------------
__global__ __launch_bounds__(256)
void k_input_gemm(const _Float16* x16, const _Float16* wih16, _Float16* xg16) {
  __shared__ __align__(16) _Float16 xs[16 * HID];      // 16-row x tile
  __shared__ __align__(16) _Float16 wchunk[G4H * 32];  // Wih[:, kk..kk+31]
  __shared__ __align__(16) _Float16 outs[16 * G4H];    // staged output tile
  int tid = threadIdx.x, wave = tid >> 5, lane = tid & 31;
  int t0 = blockIdx.x * 16;
  for (int i = tid; i < 16 * HID; i += 256) xs[i] = x16[t0 * HID + i];
  __builtin_prefetch(x16 + (t0 + 16) * HID, 0, 0);     // global_prefetch_b8
  v8f acc[4] = {};
  for (int kk = 0; kk < HID; kk += 32) {
    __syncthreads();                                   // prior wchunk readers done
    if (wave == 0) {
      // 512 rows x 32 cols strided tile of Wih, one TDM descriptor
      tdm_load_2d_f16(lds_off_u32(wchunk), wih16 + kk, 32u, 512u, 32u, 512u, 128u);
      __builtin_amdgcn_s_wait_tensorcnt(0);
    }
    __syncthreads();
    v16h a = load_a_frag(xs, HID, kk, lane);
#pragma unroll
    for (int j = 0; j < 4; ++j) {                      // 8 waves x 4 tiles = 512 cols
      v16h b = load_b_frag(wchunk, 32, (wave * 4 + j) * 16, 0, lane);
      acc[j] = wmma_f16(a, b, acc[j]);
    }
  }
#pragma unroll
  for (int j = 0; j < 4; ++j) {
    int n = (wave * 4 + j) * 16 + (lane & 15);
#pragma unroll
    for (int e = 0; e < 8; ++e) outs[cd_row(lane, e) * G4H + n] = (_Float16)acc[j][e];
  }
  __syncthreads();
  // contiguous 16KB block store (rows t0..t0+15 are adjacent in xg16)
  const uint4* src = (const uint4*)outs;
  uint4* dst = (uint4*)(xg16 + (size_t)t0 * G4H);
  for (int i = tid; i < 16 * G4H / 8; i += 256) dst[i] = src[i];
}

// ---------------- stage 3: LSTM recurrence ----------------
// Whh resident in LDS via TDM; per-step xg tile fetched with ASYNC loads that
// overlap the h @ Whh^T WMMAs; cell state c in registers; h via LDS tile.
__global__ __launch_bounds__(256)
void k_lstm(const _Float16* xg16, const _Float16* whh16,
            const int* lengths, const int* seq_off, _Float16* h16out) {
  __shared__ __align__(16) _Float16 wh[G4H * HID];   // 128 KB Whh fp16
  __shared__ __align__(16) _Float16 h_s[16 * HID];   // 4 KB hidden tile
  __shared__ __align__(16) _Float16 xg_s[16 * G4H];  // 16 KB per-step gate inputs
  __shared__ int len_s[16], off_s[16], maxlen_s;
  int tid = threadIdx.x, wave = tid >> 5, lane = tid & 31;
  int s0 = blockIdx.x * 16;
  if (tid < 16) { len_s[tid] = lengths[s0 + tid]; off_s[tid] = seq_off[s0 + tid]; }
  if (wave == 0) {
    // whole Whh as 128x512 fp16 2D tile
    tdm_load_2d_f16(lds_off_u32(wh), whh16, 128u, 512u, 128u, 512u, 128u);
  }
  for (int i = tid; i < 16 * HID; i += 256) h_s[i] = (_Float16)0.0f;
  __syncthreads();
  if (tid == 0) {
    int m = 0;
    for (int r = 0; r < 16; ++r) m = max(m, len_s[r]);
    maxlen_s = m;
  }
  if (wave == 0) __builtin_amdgcn_s_wait_tensorcnt(0);
  __syncthreads();
  int maxlen = maxlen_s;
  int ncol = wave * 16 + (lane & 15);                // this lane's hidden column
  // per-thread async-copy assignment: 4 x 16B chunks of the 16x512 xg tile
  int chunk0 = tid * 4;
  float creg[8];
#pragma unroll
  for (int e = 0; e < 8; ++e) creg[e] = 0.0f;

  for (int t = 0; t < maxlen; ++t) {
    // issue async loads of this step's 16 gate-input rows (overlaps WMMAs)
#pragma unroll
    for (int c4 = 0; c4 < 4; ++c4) {
      int c = chunk0 + c4;
      int m = c >> 6;                                 // 64 chunks per 1KB row
      int rb = (c & 63) * 16;
      unsigned voff = (unsigned)(((off_s[m] + t) * G4H) * 2 + rb);
      async_load_b128(lds_off_u32(&xg_s[m * G4H]) + rb, xg16, voff);
    }
    // recurrent GEMM: gates_h = h @ Whh^T (wave owns cols ncol of all 4 gates)
    v8f acc[4] = {};
#pragma unroll
    for (int kk = 0; kk < HID; kk += 32) {
      v16h a = load_a_frag(h_s, HID, kk, lane);
#pragma unroll
      for (int g = 0; g < 4; ++g) {
        v16h b = load_b_frag(wh, HID, g * HID + wave * 16, kk, lane);
        acc[g] = wmma_f16(a, b, acc[g]);
      }
    }
    wait_asynccnt0();
    __syncthreads();                                  // xg_s ready; h_s reads done
    // elementwise gate update entirely in-wave; c stays in registers
#pragma unroll
    for (int e = 0; e < 8; ++e) {
      int m = cd_row(lane, e);
      const _Float16* xr = &xg_s[m * G4H + ncol];
      float ig = sigmoidf_(acc[0][e] + (float)xr[0 * HID]);
      float fg = sigmoidf_(acc[1][e] + (float)xr[1 * HID]);
      float gg = tanhf    (acc[2][e] + (float)xr[2 * HID]);
      float og = sigmoidf_(acc[3][e] + (float)xr[3 * HID]);
      float c = fg * creg[e] + ig * gg;
      creg[e] = c;
      h_s[m * HID + ncol] = (_Float16)(og * tanhf(c));
    }
    __syncthreads();                                  // h_s complete
    // row-masked coalesced b128 stores of h (16 rows x 16 chunks)
    {
      int m = tid >> 4, cb = (tid & 15) * 16;
      if (t < len_s[m]) {
        uint4 v = *(const uint4*)((const char*)&h_s[m * HID] + cb);
        *(uint4*)((char*)(h16out + (off_s[m] + t) * HID) + cb) = v;
      }
    }
  }
}

// ---------------- stage 4: binned gather projection + scatter-add (WMMA + TDM) ----------------
__global__ __launch_bounds__(128)
void k_gather_proj(const _Float16* h16, const _Float16* gw16, const int* oe_idx,
                   const int* offsets, const int* counts, const int* tile_base,
                   const int* sorted_idx, float* accum) {
  __shared__ __align__(16) _Float16 hs[16 * HID];
  __shared__ __align__(16) _Float16 wsh[HID * HID];
  __shared__ int rows[16];
  int bid = blockIdx.x;
  int k = -1, tile = 0;
#pragma unroll
  for (int j = 0; j < KPOS; ++j)
    if (bid >= tile_base[j] && bid < tile_base[j + 1]) { k = j; tile = bid - tile_base[j]; }
  if (k < 0) return;
  int tid = threadIdx.x, wave = tid >> 5, lane = tid & 31;
  int cnt = counts[k], base = offsets[k];
  if (wave == 0) {
    tdm_load_2d_f16(lds_off_u32(wsh), gw16 + k * HID * HID, 16384u, 1u, 16384u, 1u, 16384u);
  }
  if (tid < 16) {
    int idx = tile * 16 + tid;
    rows[tid] = (idx < cnt) ? sorted_idx[base + idx] : -1;
  }
  __syncthreads();
  for (int i = tid; i < 16 * HID; i += 128) {
    int r = i >> 7, c = i & 127;
    int t = rows[r];
    hs[i] = (t >= 0) ? h16[t * HID + c] : (_Float16)0.0f;
  }
  if (wave == 0) __builtin_amdgcn_s_wait_tensorcnt(0);
  __syncthreads();

#pragma unroll
  for (int nt = 0; nt < 2; ++nt) {
    int n0 = (wave * 2 + nt) * 16;
    v8f acc = {};
#pragma unroll
    for (int kk = 0; kk < HID; kk += 32) {
      v16h a = load_a_frag(hs, HID, kk, lane);
      v16h b = load_b_frag(wsh, HID, n0, kk, lane);
      acc = wmma_f16(a, b, acc);
    }
    int n = n0 + (lane & 15);
#pragma unroll
    for (int e = 0; e < 8; ++e) {
      int t = rows[cd_row(lane, e)];
      if (t >= 0) atomicAdd(&accum[oe_idx[t] * HID + n], acc[e]);  // global_atomic_add_f32
    }
  }
}

// ---------------- stage 5: residual + LayerNorm ----------------
__global__ __launch_bounds__(128)
void k_ln(const float* accum, const float* events, const float* g, const float* b, float* out) {
  __shared__ float r1[128], r2[128];
  int row = blockIdx.x, c = threadIdx.x;
  float v = accum[row * HID + c] + events[row * HID + c];
  r1[c] = v; r2[c] = v * v;
  __syncthreads();
  for (int st = 64; st > 0; st >>= 1) {
    if (c < st) { r1[c] += r1[c + st]; r2[c] += r2[c + st]; }
    __syncthreads();
  }
  float mu = r1[0] * (1.0f / HID);
  float var = r2[0] * (1.0f / HID) - mu * mu;
  out[row * HID + c] = (v - mu) * rsqrtf(var + 1e-5f) * g[c] + b[c];
}

// ---------------- host launcher ----------------
extern "C" void kernel_launch(void* const* d_in, const int* in_sizes, int n_in,
                              void* d_out, int out_size, void* d_ws, size_t ws_size,
                              hipStream_t stream) {
  (void)in_sizes; (void)n_in; (void)out_size; (void)ws_size;
  const float* events = (const float*)d_in[0];
  const float* sw     = (const float*)d_in[1];
  const float* gw     = (const float*)d_in[2];
  const float* wih    = (const float*)d_in[3];
  const float* whh    = (const float*)d_in[4];
  const float* ln_g   = (const float*)d_in[5];
  const float* ln_b   = (const float*)d_in[6];
  const int* pos_id   = (const int*)d_in[7];
  const int* oe_idx   = (const int*)d_in[8];
  const int* lengths  = (const int*)d_in[9];
  float* out = (float*)d_out;

  size_t off = 0;
  char* wsb = (char*)d_ws;
  auto take = [&](size_t bytes) -> void* {
    void* p = wsb + off; off += (bytes + 255) & ~(size_t)255; return p;
  };
  _Float16* sw16  = (_Float16*)take((size_t)KPOS * HID * DIN * 2);
  _Float16* gw16  = (_Float16*)take((size_t)KPOS * HID * HID * 2);
  _Float16* wih16 = (_Float16*)take((size_t)G4H * HID * 2);
  _Float16* whh16 = (_Float16*)take((size_t)G4H * HID * 2);
  _Float16* x16   = (_Float16*)take((size_t)TTOT * HID * 2);
  _Float16* xg16  = (_Float16*)take((size_t)TTOT * G4H * 2);
  _Float16* h16   = (_Float16*)take((size_t)TTOT * HID * 2);
  float*    accum = (float*)take((size_t)NEV * HID * 4);
  int* counts     = (int*)take(KPOS * 4);
  int* offsets    = (int*)take(KPOS * 4);
  int* cursors    = (int*)take(KPOS * 4);
  int* tile_base  = (int*)take((KPOS + 1) * 4);
  int* sorted_idx = (int*)take((size_t)TTOT * 4);
  int* seq_off    = (int*)take((size_t)NSEQ * 4);

  const int nTiles = TTOT / 16 + KPOS;

  k_convert<<<(KPOS * HID * DIN + 255) / 256, 256, 0, stream>>>(
      sw, gw, wih, whh, sw16, gw16, wih16, whh16);
  k_zero_counts<<<1, 32, 0, stream>>>(counts);
  k_count<<<TTOT / 256, 256, 0, stream>>>(pos_id, counts);
  k_scan<<<1, 1, 0, stream>>>(counts, offsets, cursors, tile_base, lengths, seq_off);
  k_bin_scatter<<<TTOT / 256, 256, 0, stream>>>(pos_id, cursors, sorted_idx);
  k_zero_accum<<<NEV * HID / 256, 256, 0, stream>>>(accum);

  k_scatter_proj<<<nTiles, 128, 0, stream>>>(events, sw16, oe_idx, offsets, counts,
                                             tile_base, sorted_idx, x16);
  k_input_gemm<<<TTOT / 16, 256, 0, stream>>>(x16, wih16, xg16);
  k_lstm<<<NSEQ / 16, 256, 0, stream>>>(xg16, whh16, lengths, seq_off, h16);
  k_gather_proj<<<nTiles, 128, 0, stream>>>(h16, gw16, oe_idx, offsets, counts,
                                            tile_base, sorted_idx, accum);
  k_ln<<<NEV, 128, 0, stream>>>(accum, events, ln_g, ln_b, out);
}
